// HyperConnection_18571438588141
// MI455X (gfx1250) — matrix operand order repaired and checked by reading, not compiled, not run on hardware
//
#include <hip/hip_runtime.h>
#include <hip/hip_bf16.h>

#define B_ 2
#define S_ 4096
#define N_ 4
#define D_ 2048
#define ND_ (N_ * D_)          /* 8192 */
#define CHUNK_S 32
#define SCHUNKS (S_ / CHUNK_S) /* 128 */

typedef __attribute__((ext_vector_type(16))) _Float16 v16h;
typedef __attribute__((ext_vector_type(8)))  float    v8f;
typedef __attribute__((ext_vector_type(4)))  float    f4;

// ---------------------------------------------------------------- K0: zero accumulator
__global__ void zero_accum_kernel(float* __restrict__ acc) {
    int i = blockIdx.x * 256 + threadIdx.x;
    if (i < B_ * ND_) acc[i] = 0.0f;
}

// ---------------------------------------------------------------- K1: LayerNorm + mean-over-S
// One block handles CHUNK_S rows of one (b, n). Row = 2048 contiguous floats.
// Thread t owns d in {4t..4t+3} and {1024+4t..1024+4t+3} (two float4 per row).
__global__ __launch_bounds__(256) void ln_reduce_kernel(const float* __restrict__ H,
                                                        float* __restrict__ accum) {
    const int t    = threadIdx.x;
    const int sc   = blockIdx.x % SCHUNKS;
    const int n    = (blockIdx.x / SCHUNKS) % N_;
    const int b    = blockIdx.x / (SCHUNKS * N_);
    const int lane = t & 31;
    const int wid  = t >> 5;

    __shared__ float2 wsum[8];

    f4 acc0 = {0.f, 0.f, 0.f, 0.f};
    f4 acc1 = {0.f, 0.f, 0.f, 0.f};

    for (int s0 = 0; s0 < CHUNK_S; ++s0) {
        const int s = sc * CHUNK_S + s0;
        const f4* row = (const f4*)(H + (((size_t)b * S_ + s) * N_ + n) * D_);
        f4 x0 = row[t];
        f4 x1 = row[t + 256];

        float sm = x0.x + x0.y + x0.z + x0.w + x1.x + x1.y + x1.z + x1.w;
        float sq = x0.x * x0.x + x0.y * x0.y + x0.z * x0.z + x0.w * x0.w +
                   x1.x * x1.x + x1.y * x1.y + x1.z * x1.z + x1.w * x1.w;

        // wave32 reduction
        #pragma unroll
        for (int off = 16; off > 0; off >>= 1) {
            sm += __shfl_xor(sm, off);
            sq += __shfl_xor(sq, off);
        }
        if (lane == 0) wsum[wid] = make_float2(sm, sq);
        __syncthreads();
        float tsm = 0.f, tsq = 0.f;
        #pragma unroll
        for (int w = 0; w < 8; ++w) { tsm += wsum[w].x; tsq += wsum[w].y; }
        __syncthreads();  // protect wsum for the next row

        const float mean = tsm * (1.0f / D_);
        const float var  = tsq * (1.0f / D_) - mean * mean;
        const float rstd = rsqrtf(var + 1e-5f);

        acc0 += (x0 - mean) * rstd;
        acc1 += (x1 - mean) * rstd;
    }

    float* dst = accum + (size_t)b * ND_ + n * D_;
    const int d0 = 4 * t;
    atomicAdd(dst + d0 + 0, acc0.x);          atomicAdd(dst + d0 + 1, acc0.y);
    atomicAdd(dst + d0 + 2, acc0.z);          atomicAdd(dst + d0 + 3, acc0.w);
    atomicAdd(dst + 1024 + d0 + 0, acc1.x);   atomicAdd(dst + 1024 + d0 + 1, acc1.y);
    atomicAdd(dst + 1024 + d0 + 2, acc1.z);   atomicAdd(dst + 1024 + d0 + 3, acc1.w);
}

// ---------------------------------------------------------------- K2: deltas via WMMA + gates
// Single workgroup. H_flat finalized into f16 LDS; 24-row GEMM (K=8192) via
// v_wmma_f32_16x16x32_f16: M=16 (batch rows 0..1 live), two 16-col tiles on waves 0/1.
__global__ __launch_bounds__(256) void gates_kernel(
    const float* __restrict__ accum,
    const float* __restrict__ gamma, const float* __restrict__ beta,
    const float* __restrict__ Wpost, const float* __restrict__ Wpre, const float* __restrict__ Wres,
    const float* __restrict__ Hpost_base, const float* __restrict__ Hpre_base,
    const float* __restrict__ Hres_base,
    const float* __restrict__ s_post, const float* __restrict__ s_pre, const float* __restrict__ s_res,
    float* __restrict__ gates) {

    __shared__ _Float16 sh_hf[B_][ND_];  // 32 KB
    __shared__ float    sdelta[B_][32];

    const int t = threadIdx.x;
    // finalize H_flat = gamma * (acc/S) + beta  (gamma/beta broadcast over n)
    for (int i = t; i < B_ * ND_; i += 256) {
        const int b = i / ND_;
        const int k = i - b * ND_;
        const int d = k & (D_ - 1);
        sh_hf[b][k] = (_Float16)(accum[i] * (1.0f / S_) * gamma[d] + beta[d]);
    }
    __syncthreads();

    const int wid = t >> 5;
    if (wid < 2) {                      // waves 0,1: full EXEC inside
        const int lane = t & 31;
        const int m    = lane & 15;
        const int hi   = lane >> 4;     // half-wave selector
        const int r    = wid * 16 + m;  // global W row for B-fragment (column of D)
        const float* wrow = (r < 4)  ? (Wpost + (size_t)r * ND_)
                          : (r < 8)  ? (Wpre  + (size_t)(r - 4) * ND_)
                          : (r < 24) ? (Wres  + (size_t)(r - 8) * ND_)
                                     : nullptr;
        v8f c = {0.f, 0.f, 0.f, 0.f, 0.f, 0.f, 0.f, 0.f};

        for (int kb = 0; kb < ND_; kb += 32) {
            // A 16x32 f16: lane(0-15)=row m: K {0..7,16..23}; lane(16-31): K {8..15,24..31}
            v16h a = {};
            if (m < B_) {
                const _Float16* p0 = &sh_hf[m][kb + hi * 8];
                const _Float16* p1 = &sh_hf[m][kb + 16 + hi * 8];
                #pragma unroll
                for (int e = 0; e < 8; ++e) { a[e] = p0[e]; a[8 + e] = p1[e]; }
            }
            // B 32x16 f16: lane(0-15)=col n: K {0..7,16..23}; lane(16-31): K {8..15,24..31}
            v16h bm = {};
            if (wrow) {
                const float* q = wrow + kb + hi * 8;
                #pragma unroll
                for (int e = 0; e < 8; ++e) { bm[e] = (_Float16)q[e]; bm[8 + e] = (_Float16)q[16 + e]; }
            }
            c = __builtin_amdgcn_wmma_f32_16x16x32_f16(false, a, false, bm,
                                                       (short)0, c, false, false);
        }
        // D(m,n): VGPR m (m<8) at lane n (0-15). We need m = 0,1 (the two batches).
        if (lane < 16) {
            sdelta[0][wid * 16 + lane] = c[0];
            sdelta[1][wid * 16 + lane] = c[1];
        }
    }
    __syncthreads();

    if (t < B_) {                       // trivial scalar tail: gates + Sinkhorn
        const int b = t;
        const float sp = *s_post, spr = *s_pre, sr = *s_res;
        float post[4], pre[4], M[4][4];
        float presum = 0.f;
        #pragma unroll
        for (int n = 0; n < 4; ++n) {
            post[n] = 1.0f / (1.0f + expf(-(Hpost_base[n] + sdelta[b][n] * sp)));
            const float pv = 1.0f / (1.0f + expf(-(Hpre_base[n] + sdelta[b][4 + n] * spr)));
            pre[n] = pv; presum += pv;
        }
        const float pinv = 1.0f / (presum + 1e-8f);
        #pragma unroll
        for (int n = 0; n < 4; ++n) pre[n] *= pinv;
        for (int i = 0; i < 4; ++i)
            for (int j = 0; j < 4; ++j)
                M[i][j] = expf(Hres_base[i * 4 + j] + sdelta[b][8 + i * 4 + j] * sr);
        for (int it = 0; it < 20; ++it) {
            for (int i = 0; i < 4; ++i) {
                const float inv = 1.0f / (M[i][0] + M[i][1] + M[i][2] + M[i][3] + 1e-8f);
                for (int j = 0; j < 4; ++j) M[i][j] *= inv;
            }
            for (int j = 0; j < 4; ++j) {
                const float inv = 1.0f / (M[0][j] + M[1][j] + M[2][j] + M[3][j] + 1e-8f);
                for (int i = 0; i < 4; ++i) M[i][j] *= inv;
            }
        }
        float* g = gates + b * 24;
        for (int n = 0; n < 4; ++n) { g[n] = post[n]; g[4 + n] = pre[n]; }
        for (int i = 0; i < 4; ++i)
            for (int j = 0; j < 4; ++j) g[8 + i * 4 + j] = M[i][j];
    }
}

// ---------------------------------------------------------------- K3: fused mixing pass
// Per (b,s,d4): read 4 H rows + branch_output, write branch_input + 4 H_new rows.
// All big traffic is last-use / streaming -> non-temporal loads and stores
// (TH=NT) so this pass doesn't thrash the 192 MB L2.
__global__ __launch_bounds__(256) void mix_kernel(const float* __restrict__ H,
                                                  const float* __restrict__ BO,
                                                  const float* __restrict__ gates,
                                                  float* __restrict__ out_branch,
                                                  float* __restrict__ out_hnew) {
    const unsigned gid = blockIdx.x * 256u + threadIdx.x;
    const int D4 = D_ / 4;  // 512
    const int d4 = gid % D4;
    const int s  = (gid / D4) % S_;
    const int b  = gid / (unsigned)(D4 * S_);

    const float* g = gates + b * 24;
    float post[4], pre[4], R[16];
    #pragma unroll
    for (int i = 0; i < 4; ++i) { post[i] = g[i]; pre[i] = g[4 + i]; }
    #pragma unroll
    for (int i = 0; i < 16; ++i) R[i] = g[8 + i];

    const f4* H4  = (const f4*)H;
    const f4* BO4 = (const f4*)BO;
    f4* OB4 = (f4*)out_branch;
    f4* ON4 = (f4*)out_hnew;

    const size_t base = ((size_t)b * S_ + s) * N_ * D4;
    f4 h[4];
    #pragma unroll
    for (int j = 0; j < 4; ++j)
        h[j] = __builtin_nontemporal_load(&H4[base + (size_t)j * D4 + d4]);
    const f4 bo = __builtin_nontemporal_load(&BO4[((size_t)b * S_ + s) * D4 + d4]);

    const f4 bi = pre[0] * h[0] + pre[1] * h[1] + pre[2] * h[2] + pre[3] * h[3];
    __builtin_nontemporal_store(bi, &OB4[((size_t)b * S_ + s) * D4 + d4]);

    #pragma unroll
    for (int i = 0; i < 4; ++i) {
        const f4 o = R[i*4+0] * h[0] + R[i*4+1] * h[1] +
                     R[i*4+2] * h[2] + R[i*4+3] * h[3] + post[i] * bo;
        __builtin_nontemporal_store(o, &ON4[base + (size_t)i * D4 + d4]);
    }
}

// ---------------------------------------------------------------- launch
extern "C" void kernel_launch(void* const* d_in, const int* in_sizes, int n_in,
                              void* d_out, int out_size, void* d_ws, size_t ws_size,
                              hipStream_t stream) {
    const float* H          = (const float*)d_in[0];
    const float* BO         = (const float*)d_in[1];
    const float* gamma      = (const float*)d_in[2];
    const float* beta       = (const float*)d_in[3];
    const float* Hpost_base = (const float*)d_in[4];
    const float* Hpre_base  = (const float*)d_in[5];
    const float* Hres_base  = (const float*)d_in[6];
    const float* Wpost      = (const float*)d_in[7];
    const float* Wpre       = (const float*)d_in[8];
    const float* Wres       = (const float*)d_in[9];
    const float* s_post     = (const float*)d_in[10];
    const float* s_pre      = (const float*)d_in[11];
    const float* s_res      = (const float*)d_in[12];

    float* accum = (float*)d_ws;            // B*ND floats
    float* gates = accum + B_ * ND_;        // 48 floats
    float* out_branch = (float*)d_out;                                   // B*S*D
    float* out_hnew   = out_branch + (size_t)B_ * S_ * D_;               // B*S*N*D

    zero_accum_kernel<<<(B_ * ND_ + 255) / 256, 256, 0, stream>>>(accum);
    ln_reduce_kernel<<<B_ * N_ * SCHUNKS, 256, 0, stream>>>(H, accum);
    gates_kernel<<<1, 256, 0, stream>>>(accum, gamma, beta, Wpost, Wpre, Wres,
                                        Hpost_base, Hpre_base, Hres_base,
                                        s_post, s_pre, s_res, gates);
    const unsigned mix_threads = (unsigned)B_ * S_ * (D_ / 4);  // 4,194,304
    mix_kernel<<<mix_threads / 256, 256, 0, stream>>>(H, BO, gates, out_branch, out_hnew);
}
